// MultiFrameModelRecOnly_46660524703817
// MI455X (gfx1250) — matrix operand order repaired and checked by reading, not compiled, not run on hardware
//
#include <hip/hip_runtime.h>

// ---------------------------------------------------------------------------
// Types / helpers
// ---------------------------------------------------------------------------
typedef __attribute__((ext_vector_type(16))) __bf16        v16bf;
typedef __attribute__((ext_vector_type(8)))  float         v8f;
typedef __attribute__((ext_vector_type(4)))  unsigned int  v4u;

union Frag16 { v16bf v; unsigned short u[16]; v4u q[2]; };

__device__ __forceinline__ unsigned short f2bf(float f) {
  unsigned int x = __float_as_uint(f);
  return (unsigned short)((x + 0x7FFFu + ((x >> 16) & 1u)) >> 16);  // RNE
}
__device__ __forceinline__ float bf2f(unsigned short h) {
  return __uint_as_float(((unsigned int)h) << 16);
}
__device__ __forceinline__ float sigmoidf_(float x) {
  return 1.0f / (1.0f + __expf(-x));
}

// ---------------------------------------------------------------------------
// Conversion kernels. Weights are packed into WMMA *fragment-major* layout:
// for each (k-tile of 32) x (n-tile of 16), store the per-lane 16-half B
// fragment contiguously: dst[tile*512 + lane*16 + i] where the element is
// W[k0 + (i&7) + ((lane&16)>>1) + ((i&8)<<1)][n0 + (lane&15)].  B-fragment loads then become two
// coalesced b128 loads per lane.
// ---------------------------------------------------------------------------
__global__ void cvt_f32_to_bf16(const float* __restrict__ src,
                                unsigned short* __restrict__ dst, int n) {
  int i = blockIdx.x * blockDim.x + threadIdx.x;
  if (i < n) dst[i] = f2bf(src[i]);
}

// Input-projection weights: src [K, 3U] (gate order z|r|h) -> packed
// [Kpad/32][3*Upad/16][512], gate segments padded to Upad.
__global__ void cvt_weight_gemm(const float* __restrict__ src, unsigned short* __restrict__ dst,
                                int K, int U, int Kpad, int Upad) {
  int idx = blockIdx.x * blockDim.x + threadIdx.x;
  int ntN = (3 * Upad) >> 4;
  int total = (Kpad >> 5) * ntN * 512;
  if (idx >= total) return;
  int tile = idx >> 9, within = idx & 511;
  int lane = within >> 4, i = within & 15;
  int kt = tile / ntN, nj = tile - kt * ntN;
  int kk = kt * 32 + (i & 7) + ((lane & 16) >> 1) + ((i & 8) << 1);
  int c = nj * 16 + (lane & 15);
  int g = c / Upad, uu = c - g * Upad;
  float v = (kk < K && uu < U) ? src[(size_t)kk * (3 * U) + g * U + uu] : 0.0f;
  dst[idx] = f2bf(v);
}

// Recurrent weights: src [U, 3U] -> packed [Upad/32][Upad/16][3 gates][512].
__global__ void cvt_weight_rec(const float* __restrict__ src, unsigned short* __restrict__ dst,
                               int U, int Upad) {
  int idx = blockIdx.x * blockDim.x + threadIdx.x;
  int nU = Upad >> 4;
  int total = (Upad >> 5) * nU * 3 * 512;
  if (idx >= total) return;
  int tile = idx >> 9, within = idx & 511;
  int lane = within >> 4, i = within & 15;
  int g = tile % 3, t2 = tile / 3;
  int uj = t2 % nU, kt = t2 / nU;
  int kk = kt * 32 + (i & 7) + ((lane & 16) >> 1) + ((i & 8) << 1);
  int col = uj * 16 + (lane & 15);
  float v = (kk < U && col < U) ? src[(size_t)kk * (3 * U) + g * U + col] : 0.0f;
  dst[idx] = f2bf(v);
}

// src: [2, 3U] -> dst_i [3*Upad], dst_r [3*Upad] (fp32, zero padded)
__global__ void cvt_bias(const float* __restrict__ src, float* __restrict__ dst_i,
                         float* __restrict__ dst_r, int U, int Upad) {
  int idx = blockIdx.x * blockDim.x + threadIdx.x;
  int X3 = 3 * Upad;
  if (idx >= X3) return;
  int g = idx / Upad, uu = idx - g * Upad;
  dst_i[idx] = (uu < U) ? src[g * U + uu] : 0.0f;
  dst_r[idx] = (uu < U) ? src[3 * U + g * U + uu] : 0.0f;
}

// ---------------------------------------------------------------------------
// Input projection: C[M,N] = A[M,K](bf16, ld=lda) * Wp(packed bf16) + bias[N]
// Each wave computes two adjacent 16x16 tiles (shared A fragment).
// A fragment = 2x global b128; each B fragment = 2x global b128 (packed).
// ---------------------------------------------------------------------------
__global__ void __launch_bounds__(256)
gemm_bias_bf16(const unsigned short* __restrict__ A, int lda,
               const unsigned short* __restrict__ Wp, const float* __restrict__ bias,
               float* __restrict__ C, int M, int N, int K) {
  const int lane = threadIdx.x & 31;
  const int wv = blockIdx.x * (blockDim.x >> 5) + (threadIdx.x >> 5);
  const int ntN = N >> 4;
  const int ngrp = ntN >> 1;                 // pairs of column tiles
  const int total = (M >> 4) * ngrp;
  if (wv >= total) return;                   // wave-uniform
  const int mi = wv / ngrp, gj = wv - mi * ngrp;
  const int nj0 = gj << 1;
  const int m0 = mi << 4;
  const int row = m0 + (lane & 15);
  const int colL = lane & 15;
  const int col0 = nj0 * 16 + colL;
  const float b0 = bias[col0], b1 = bias[col0 + 16];
  v8f acc0, acc1;
#pragma unroll
  for (int v = 0; v < 8; ++v) { acc0[v] = b0; acc1[v] = b1; }
  const unsigned short* Arow = A + (size_t)row * lda;
  const int aoff = (lane & 16) >> 1;
  for (int k0 = 0; k0 < K; k0 += 32) {
    Frag16 a, bf0, bf1;
    const v4u* Ap = (const v4u*)(Arow + k0 + aoff);
    a.q[0] = Ap[0];                      // K chunk [k0+aoff .. +7]
    a.q[1] = Ap[2];                      // K chunk [k0+16+aoff .. +7]
    const unsigned short* Bbase =
        Wp + (((size_t)(k0 >> 5) * ntN + nj0) << 9) + (lane << 4);
    const v4u* Bp0 = (const v4u*)Bbase;
    bf0.q[0] = Bp0[0]; bf0.q[1] = Bp0[1];
    const v4u* Bp1 = (const v4u*)(Bbase + 512);
    bf1.q[0] = Bp1[0]; bf1.q[1] = Bp1[1];
    __builtin_prefetch(Bbase + ((size_t)ntN << 9), 0, 1);  // next k-tile
    acc0 = __builtin_amdgcn_wmma_f32_16x16x32_bf16(false, a.v, false, bf0.v,
                                                   (short)0, acc0, false, false);
    acc1 = __builtin_amdgcn_wmma_f32_16x16x32_bf16(false, a.v, false, bf1.v,
                                                   (short)0, acc1, false, false);
  }
  const int crow = m0 + ((lane & 16) >> 1);
#pragma unroll
  for (int v = 0; v < 8; ++v) {
    C[(size_t)(crow + v) * N + col0] = acc0[v];
    C[(size_t)(crow + v) * N + col0 + 16] = acc1[v];
  }
}

// ---------------------------------------------------------------------------
// Persistent GRU scan: one workgroup, h[32,UPAD] bf16 resident in LDS.
// Per timestep each wave owns 16x16 unit tiles: 3 gate WMMAs (z/r/h) sharing
// one A fragment read from LDS via 2x ds_load_b128; R read via packed b128.
// Gate math in registers, barrier, h write-back, barrier.
// ---------------------------------------------------------------------------
template <int UPAD, int NW, int TPW, int ACT>
__global__ void __launch_bounds__(NW * 32)
gru_recurrent(const float* __restrict__ xp,          // [(b*512+t), 3*UPAD]
              const unsigned short* __restrict__ Rp, // packed recurrent weights
              const float* __restrict__ rb,          // [3*UPAD]
              unsigned short* __restrict__ outb, int out_ld, int out_col,
              float* __restrict__ outf, int u_real) {
  constexpr int X3 = 3 * UPAD;
  constexpr int NU = UPAD >> 4;            // unit tiles
  constexpr int NTILES = 2 * NU;           // B=32 -> 2 row tiles
  extern __shared__ unsigned short hsm[];  // [32][UPAD] bf16
  const int tid = threadIdx.x;
  const int lane = tid & 31;
  const int wv = tid >> 5;
  for (int i = tid; i < 32 * UPAD; i += NW * 32) hsm[i] = 0;
  __syncthreads();
  const int aoff = (lane & 16) >> 1;

  for (int t = 0; t < 512; ++t) {
    float hnew[TPW][8];
#pragma unroll
    for (int rep = 0; rep < TPW; ++rep) {
      const int tile = wv + rep * NW;
      if (tile < NTILES) {
        const int mi = tile & 1, uj = tile >> 1;
        const int hrow = mi * 16 + (lane & 15);
        const int col = uj * 16 + (lane & 15);
        v8f az, ar, ah;
#pragma unroll
        for (int v = 0; v < 8; ++v) { az[v] = 0.f; ar[v] = 0.f; ah[v] = 0.f; }
        for (int k0 = 0; k0 < UPAD; k0 += 32) {
          Frag16 a, bz, brf, bh;
          const v4u* Hp = (const v4u*)(hsm + hrow * UPAD + k0 + aoff);
          a.q[0] = Hp[0];                  // ds_load_b128
          a.q[1] = Hp[2];
          const unsigned short* Rbase =
              Rp + ((((size_t)(k0 >> 5) * NU + uj) * 3) << 9) + (lane << 4);
          const v4u* Rz = (const v4u*)Rbase;
          bz.q[0] = Rz[0];  bz.q[1] = Rz[1];
          const v4u* Rr = (const v4u*)(Rbase + 512);
          brf.q[0] = Rr[0]; brf.q[1] = Rr[1];
          const v4u* Rh = (const v4u*)(Rbase + 1024);
          bh.q[0] = Rh[0];  bh.q[1] = Rh[1];
          az = __builtin_amdgcn_wmma_f32_16x16x32_bf16(false, a.v, false, bz.v,
                                                       (short)0, az, false, false);
          ar = __builtin_amdgcn_wmma_f32_16x16x32_bf16(false, a.v, false, brf.v,
                                                       (short)0, ar, false, false);
          ah = __builtin_amdgcn_wmma_f32_16x16x32_bf16(false, a.v, false, bh.v,
                                                       (short)0, ah, false, false);
        }
        // reset_after gates, entirely in registers
#pragma unroll
        for (int v = 0; v < 8; ++v) {
          const int brow = mi * 16 + v + ((lane & 16) >> 1);  // batch index
          const int u = col;
          const float* xrow = xp + (size_t)(brow * 512 + t) * X3;
          const float z = sigmoidf_(xrow[u] + az[v] + rb[u]);
          const float r = sigmoidf_(xrow[UPAD + u] + ar[v] + rb[UPAD + u]);
          const float pre = xrow[2 * UPAD + u] + r * (ah[v] + rb[2 * UPAD + u]);
          const float hh = (ACT == 0) ? fmaxf(pre, 0.0f) : pre;
          const float hold = bf2f(hsm[brow * UPAD + u]);
          hnew[rep][v] = z * hold + (1.0f - z) * hh;
        }
      }
    }
    __syncthreads();  // all reads of h done before any write
#pragma unroll
    for (int rep = 0; rep < TPW; ++rep) {
      const int tile = wv + rep * NW;
      if (tile < NTILES) {
        const int mi = tile & 1, uj = tile >> 1;
        const int u = uj * 16 + (lane & 15);
#pragma unroll
        for (int v = 0; v < 8; ++v) {
          const int brow = mi * 16 + v + ((lane & 16) >> 1);
          const unsigned short hb = f2bf(hnew[rep][v]);
          hsm[brow * UPAD + u] = hb;  // ds_store_b16
          if (outb) outb[(size_t)(brow * 512 + t) * out_ld + out_col + u] = hb;
          if (outf && u < u_real) outf[(size_t)(brow * 512 + t) * u_real + u] = hnew[rep][v];
        }
      }
    }
    __syncthreads();  // h visible for next step
  }
}

// ---------------------------------------------------------------------------
// Host launcher
// ---------------------------------------------------------------------------
extern "C" void kernel_launch(void* const* d_in, const int* in_sizes, int n_in,
                              void* d_out, int out_size, void* d_ws, size_t ws_size,
                              hipStream_t stream) {
  (void)in_sizes; (void)n_in; (void)out_size; (void)ws_size;
  const float* x = (const float*)d_in[0];
  const float *k[6], *r[6], *b[6];
  for (int i = 0; i < 6; ++i) {
    k[i] = (const float*)d_in[1 + 3 * i];
    r[i] = (const float*)d_in[2 + 3 * i];
    b[i] = (const float*)d_in[3 + 3 * i];
  }
  // original dims / padded dims (gate segments padded to multiples of 32)
  const int Korig[6] = {512, 1024, 128, 151, 256, 128};
  const int Uorig[6] = {1024, 128, 23, 256, 128, 8};
  const int Upad[6]  = {1024, 128, 32, 256, 128, 32};
  const int Kpad[6]  = {512, 1024, 128, 160, 256, 128};
  const size_t M = 32 * 512;  // B*T rows, row = b*T + t

  char* base = (char*)d_ws;
  size_t off = 0;
  auto alloc = [&](size_t bytes) -> void* {
    void* p = base + off;
    off = (off + bytes + 255) & ~(size_t)255;
    return p;
  };
  unsigned short* xbf  = (unsigned short*)alloc(M * 512 * 2);
  unsigned short* out1 = (unsigned short*)alloc(M * 1024 * 2);
  unsigned short* yc   = (unsigned short*)alloc(M * 160 * 2);  // [z1(128) | ye(32pad)]
  unsigned short* out4 = (unsigned short*)alloc(M * 256 * 2);
  unsigned short* out5 = (unsigned short*)alloc(M * 128 * 2);
  float* xp            = (float*)alloc(M * 3072 * 4);          // reused per layer
  unsigned short* wK[6]; unsigned short* wR[6]; float* bi[6]; float* br[6];
  for (int i = 0; i < 6; ++i) {
    wK[i] = (unsigned short*)alloc((size_t)Kpad[i] * 3 * Upad[i] * 2);
    wR[i] = (unsigned short*)alloc((size_t)Upad[i] * 3 * Upad[i] * 2);
    bi[i] = (float*)alloc((size_t)3 * Upad[i] * 4);
    br[i] = (float*)alloc((size_t)3 * Upad[i] * 4);
  }

  // --- weight / input conversion (fragment-major packing) ---
  { int n = (int)(M * 512);
    cvt_f32_to_bf16<<<(n + 255) / 256, 256, 0, stream>>>(x, xbf, n); }
  for (int i = 0; i < 6; ++i) {
    int tk = Kpad[i] * 3 * Upad[i];
    cvt_weight_gemm<<<(tk + 255) / 256, 256, 0, stream>>>(k[i], wK[i], Korig[i], Uorig[i],
                                                          Kpad[i], Upad[i]);
    int tr = Upad[i] * 3 * Upad[i];
    cvt_weight_rec<<<(tr + 255) / 256, 256, 0, stream>>>(r[i], wR[i], Uorig[i], Upad[i]);
    int tb = 3 * Upad[i];
    cvt_bias<<<(tb + 255) / 256, 256, 0, stream>>>(b[i], bi[i], br[i], Uorig[i], Upad[i]);
  }

  float* ye = (float*)d_out;                 // [B,T,23]
  float* yd = ye + M * 23;                   // [B,T,8]

  auto gemm = [&](const unsigned short* A, int lda, int i) {
    const int N = 3 * Upad[i];
    const int total = (int)(M / 16) * ((N / 16) / 2);  // 2 col-tiles per wave
    const int blocks = (total + 7) / 8;                // 8 waves / block
    gemm_bias_bf16<<<blocks, 256, 0, stream>>>(A, lda, wK[i], bi[i], xp, (int)M, N, Kpad[i]);
  };

  // L1: 512 -> 1024, relu
  gemm(xbf, 512, 0);
  gru_recurrent<1024, 16, 8, 0><<<1, 512, 32 * 1024 * 2, stream>>>(
      xp, wR[0], br[0], out1, 1024, 0, (float*)nullptr, 0);
  // L2: 1024 -> 128, relu (writes z1 into yc[:,0:128])
  gemm(out1, 1024, 1);
  gru_recurrent<128, 16, 1, 0><<<1, 512, 32 * 128 * 2, stream>>>(
      xp, wR[1], br[1], yc, 160, 0, (float*)nullptr, 0);
  // L3: 128 -> 23, linear (reads yc[:,0:128]; writes ye into yc[:,128:160] + d_out)
  gemm(yc, 160, 2);
  gru_recurrent<32, 16, 1, 1><<<1, 512, 32 * 32 * 2, stream>>>(
      xp, wR[2], br[2], yc, 160, 128, ye, 23);
  // L4: 151(pad 160) -> 256, relu (reads full yc = concat)
  gemm(yc, 160, 3);
  gru_recurrent<256, 16, 2, 0><<<1, 512, 32 * 256 * 2, stream>>>(
      xp, wR[3], br[3], out4, 256, 0, (float*)nullptr, 0);
  // L5: 256 -> 128, relu
  gemm(out4, 256, 4);
  gru_recurrent<128, 16, 1, 0><<<1, 512, 32 * 128 * 2, stream>>>(
      xp, wR[4], br[4], out5, 128, 0, (float*)nullptr, 0);
  // L6: 128 -> 8, linear -> yd
  gemm(out5, 128, 5);
  gru_recurrent<32, 16, 1, 1><<<1, 512, 32 * 32 * 2, stream>>>(
      xp, wR[5], br[5], (unsigned short*)nullptr, 0, 0, yd, 8);
}
